// Correlation_12713103196781
// MI455X (gfx1250) — matrix operand order repaired and checked by reading, not compiled, not run on hardware
//
#include <hip/hip_runtime.h>
#include <hip/hip_bf16.h>

typedef __attribute__((ext_vector_type(16))) __bf16 v16bf;
typedef __attribute__((ext_vector_type(8)))  __bf16 v8bf;
typedef __attribute__((ext_vector_type(8)))  float  v8f;

#define B_   16
#define C_   256
#define H_   48
#define W_   64
#define ND   21      // displacements per axis (441 total)
#define LDK  264     // padded per-pixel channel stride in LDS (bf16 elems, 528B: 16B-aligned, bank-skewed)

// Load a 16x32 bf16 WMMA operand fragment from a channel-contiguous LDS row.
// A-matrix layout: lane group g holds K = g*16 .. g*16+15 (contiguous)
// B-matrix layout: lane group g holds K = g*16 .. g*16+15 (contiguous)
static __device__ __forceinline__ v16bf load_frag(const __bf16* p, int off2) {
    v8bf lo = *(const v8bf*)(p);
    v8bf hi = *(const v8bf*)(p + off2);
    v16bf f;
    #pragma unroll
    for (int i = 0; i < 8; ++i) { f[i] = lo[i]; f[8 + i] = hi[i]; }
    return f;
}

// One workgroup = one (batch, output-row h). 4 waves (128 threads):
//   wave -> (parity p, A-tile u0).  Each wave owns 3 banded 16x16 output tiles.
__global__ __launch_bounds__(128)
void corr_wmma_bf16_kernel(const float* __restrict__ in1,
                           const float* __restrict__ in2,
                           float* __restrict__ out)
{
    __shared__ __bf16 a_lds[2][32][LDK];   // in1 row, parity-split: [p][u][c]
    __shared__ __bf16 b_lds[2][64][LDK];   // in2 row, parity-split + band padding: [p][u'+16][c]
    __shared__ float  o_lds[ND][W_];       // per-dy output staging for coalesced stores

    const int tid = threadIdx.x;
    const int bb  = blockIdx.x / H_;
    const int h   = blockIdx.x % H_;

    // ---- stage input1 row h: fp32 -> bf16, parity split (float4 loads, all chunks in-bounds) ----
    for (int idx = tid; idx < C_ * (W_ / 4); idx += 128) {
        int c  = idx >> 4;
        int w0 = (idx & 15) << 2;                       // coalesced over w
        float4 v = *(const float4*)&in1[(((size_t)bb * C_ + c) * H_ + h) * W_ + w0];
        int u = w0 >> 1;
        a_lds[0][u    ][c] = (__bf16)v.x;
        a_lds[1][u    ][c] = (__bf16)v.y;
        a_lds[0][u + 1][c] = (__bf16)v.z;
        a_lds[1][u + 1][c] = (__bf16)v.w;
    }
    __syncthreads();

    const int lane = tid & 31;                // wave32
    const int wv   = tid >> 5;
    const int p    = wv >> 1;                 // pixel parity handled by this wave
    const int u0   = (wv & 1) << 4;           // A-tile row base (u index)
    const int lrow = lane & 15;
    const int lg   = lane >> 4;               // lane group (0 or 1)
    const float inv = 1.0f / (float)C_;

    for (int dy = 0; dy < ND; ++dy) {
        const int row2 = h + 2 * dy - 20;
        const bool rowok = (row2 >= 0) & (row2 < H_);

        // ---- stage input2 row row2, padded window w' in [-32,96) ----
        // Chunks of 4 are aligned to the window start, so each float4 chunk is
        // either fully inside [0,W) or fully zero: no partial chunks.
        for (int idx = tid; idx < C_ * 32; idx += 128) {
            int c   = idx >> 5;
            int wp0 = ((idx & 31) << 2) - 32;           // chunk start w' (even)
            float4 v = {0.f, 0.f, 0.f, 0.f};
            if (rowok && (unsigned)wp0 < (unsigned)W_)
                v = *(const float4*)&in2[(((size_t)bb * C_ + c) * H_ + row2) * W_ + wp0];
            int s2 = (wp0 >> 1) + 16;                   // u' + 16 in [0,64)
            b_lds[0][s2    ][c] = (__bf16)v.x;
            b_lds[1][s2    ][c] = (__bf16)v.y;
            b_lds[0][s2 + 1][c] = (__bf16)v.z;
            b_lds[1][s2 + 1][c] = (__bf16)v.w;
        }
        // prefetch next dy's in2 row while it is still far away (global_prefetch_b8)
        if (dy + 1 < ND) {
            int nrow = h + 2 * (dy + 1) - 20;
            if (nrow >= 0 && nrow < H_) {
                const float* pf = in2 + (((size_t)bb * C_ + (tid << 1)) * H_ + nrow) * W_;
                __builtin_prefetch(pf, 0, 0);
            }
        }
        __syncthreads();

        // ---- banded GEMM: A tile (16u x 256c) x three B tiles (256c x 16u') ----
        v8f acc[3] = {};
        #pragma unroll
        for (int k0 = 0; k0 < C_; k0 += 32) {
            // Load ALL fragments of this K-step first, then issue the 3 WMMAs,
            // so only one dscnt wait gates the matrix ops.
            v16bf afrag = load_frag(&a_lds[p][u0 + lrow][k0 + (lg << 4)], 8);
            v16bf bf0   = load_frag(&b_lds[p][u0      + lrow][k0 + (lg << 4)], 8);
            v16bf bf1   = load_frag(&b_lds[p][u0 + 16 + lrow][k0 + (lg << 4)], 8);
            v16bf bf2   = load_frag(&b_lds[p][u0 + 32 + lrow][k0 + (lg << 4)], 8);
            acc[0] = __builtin_amdgcn_wmma_f32_16x16x32_bf16(false, afrag, false, bf0,
                                                             (short)0, acc[0], false, false);
            acc[1] = __builtin_amdgcn_wmma_f32_16x16x32_bf16(false, afrag, false, bf1,
                                                             (short)0, acc[1], false, false);
            acc[2] = __builtin_amdgcn_wmma_f32_16x16x32_bf16(false, afrag, false, bf2,
                                                             (short)0, acc[2], false, false);
        }

        // ---- band extraction: keep only |e| <= 10 diagonals, scale by 1/C ----
        #pragma unroll
        for (int j = 0; j < 3; ++j) {
            #pragma unroll
            for (int r = 0; r < 8; ++r) {
                int M  = r + (lg << 3);         // C/D layout: lanes>=16 hold M = r+8
                int N  = lane & 15;
                int e  = 16 * j - 16 + N - M;   // u' - u
                int dx = e + 10;
                if ((unsigned)dx < (unsigned)ND) {
                    int u = u0 + M;
                    o_lds[dx][2 * u + p] = acc[j][r] * inv;
                }
            }
        }
        __syncthreads();

        // ---- coalesced b128 stores: 21 dx-planes, 64 contiguous floats each ----
        for (int idx = tid; idx < ND * (W_ / 4); idx += 128) {
            int dx = idx >> 4;
            int w0 = (idx & 15) << 2;
            float4 v = *(const float4*)&o_lds[dx][w0];
            *(float4*)&out[(((size_t)bb * (ND * ND) + dy * ND + dx) * H_ + h) * W_ + w0] = v;
        }
        __syncthreads();   // protect b_lds / o_lds reuse next iteration
    }
}

extern "C" void kernel_launch(void* const* d_in, const int* in_sizes, int n_in,
                              void* d_out, int out_size, void* d_ws, size_t ws_size,
                              hipStream_t stream)
{
    (void)in_sizes; (void)n_in; (void)out_size; (void)d_ws; (void)ws_size;
    const float* in1 = (const float*)d_in[0];
    const float* in2 = (const float*)d_in[1];
    float* out = (float*)d_out;
    dim3 grid(B_ * H_), block(128);
    hipLaunchKernelGGL(corr_wmma_bf16_kernel, grid, block, 0, stream, in1, in2, out);
}
